// LSTMEncoder_65223373357507
// MI455X (gfx1250) — compile-verified, matrix operand order checked by reading, not run on hardware
//
#include <hip/hip_runtime.h>

typedef __attribute__((ext_vector_type(16))) _Float16 v16h;
typedef __attribute__((ext_vector_type(8)))  _Float16 v8h;
typedef __attribute__((ext_vector_type(8)))  float    v8f;
typedef __attribute__((ext_vector_type(4)))  float    v4f;

#define B_   512
#define T_   1000
#define I_   64
#define H_   256
#define G4_  1024   // 4*H
#define OUT_ 128

__device__ __forceinline__ float sigf(float x) { return 1.0f / (1.0f + __expf(-x)); }
__device__ __forceinline__ float tanhfast(float x) {
    float e = __expf(2.0f * x);
    return (e - 1.0f) / (e + 1.0f);
}
__device__ __forceinline__ v8f wmma_f16(v16h a, v16h b, v8f c) {
    // v_wmma_f32_16x16x32_f16: D = A*B + C, f32 accumulate
    return __builtin_amdgcn_wmma_f32_16x16x32_f16(false, a, false, b, (short)0, c,
                                                  false, false);
}

// ---------------------------------------------------------------------------
// One-shot init: zero c and h0, build combined bias.
// ---------------------------------------------------------------------------
__global__ __launch_bounds__(256) void init_state(float* __restrict__ c,
                                                  _Float16* __restrict__ h0,
                                                  const float* __restrict__ b_ih,
                                                  const float* __restrict__ b_hh,
                                                  float* __restrict__ bsum) {
    int i = blockIdx.x * blockDim.x + threadIdx.x;
    if (i < B_ * H_) {
        c[i]  = 0.0f;
        h0[i] = (_Float16)0.0f;
    }
    if (i < G4_) bsum[i] = b_ih[i] + b_hh[i];
}

// ---------------------------------------------------------------------------
// One-shot weight swizzle: pack [Wih^T ; Whh^T] (320 x 1024) f32 into f16
// B-fragment order: idx = ((kc*64 + ntile)*32 + lane)*16 + p
//   lane<16: col = lane,    halves p<8 -> K = kc*32 + p,      p>=8 -> K = kc*32+16+(p-8)
//   lane>=16: col = lane-16, halves p<8 -> K = kc*32 + 8 + p, p>=8 -> K = kc*32+24+(p-8)
// ---------------------------------------------------------------------------
__global__ __launch_bounds__(256) void swizzle_w(const float* __restrict__ w_ih,
                                                 const float* __restrict__ w_hh,
                                                 _Float16* __restrict__ Wswz) {
    int idx  = blockIdx.x * blockDim.x + threadIdx.x;   // 0 .. 327679
    int p    = idx & 15;
    int lane = (idx >> 4) & 31;
    int nt   = (idx >> 9) & 63;
    int kc   = idx >> 15;                               // 0..9
    int kb   = (lane < 16) ? 0 : 8;
    int kl   = (p < 8) ? (kb + p) : (kb + 16 + (p - 8));
    int K    = kc * 32 + kl;                            // 0..319
    int N    = nt * 16 + (lane & 15);                   // 0..1023
    float v  = (K < I_) ? w_ih[(size_t)N * I_ + K]
                        : w_hh[(size_t)N * H_ + (K - I_)];
    Wswz[idx] = (_Float16)v;
}

// ---------------------------------------------------------------------------
// Fused LSTM step: one wave owns a 16(batch) x 16(hidden) patch and computes
// all 4 gate tiles (i,f,g,o) via WMMA, then the cell update in-register.
// Grid: 128 blocks x 128 threads = 512 waves = 32 m-tiles x 16 j-tiles.
// ---------------------------------------------------------------------------
__global__ __launch_bounds__(128) void lstm_step(const float* __restrict__ x,
                                                 const _Float16* __restrict__ Wswz,
                                                 const float* __restrict__ bsum,
                                                 const _Float16* __restrict__ h_in,
                                                 _Float16* __restrict__ h_out,
                                                 float* __restrict__ c,
                                                 int t) {
    const int lane = threadIdx.x & 31;
    const int wid  = (blockIdx.x * blockDim.x + threadIdx.x) >> 5;  // 0..511
    const int mt   = wid & 31;          // batch tile
    const int jt   = wid >> 5;          // hidden tile 0..15
    const int m0   = mt * 16;
    const int j0   = jt * 16;
    const int row  = m0 + (lane & 15);  // A-matrix row for this lane
    const int kb   = (lane < 16) ? 0 : 8;

    v8f acc0 = {}, acc1 = {}, acc2 = {}, acc3 = {};

    // ---- x contribution: K chunks 0..1 (K = 0..63), f32 -> f16 on the fly
    const float* xrow = x + ((size_t)row * T_ + t) * I_;
#pragma unroll
    for (int kc = 0; kc < 2; ++kc) {
        const int K0 = kc * 32 + kb;
        v4f x0 = *(const v4f*)(xrow + K0);
        v4f x1 = *(const v4f*)(xrow + K0 + 4);
        v4f x2 = *(const v4f*)(xrow + K0 + 16);
        v4f x3 = *(const v4f*)(xrow + K0 + 20);
        v16h a;
#pragma unroll
        for (int i = 0; i < 4; ++i) {
            a[i]      = (_Float16)x0[i];
            a[4 + i]  = (_Float16)x1[i];
            a[8 + i]  = (_Float16)x2[i];
            a[12 + i] = (_Float16)x3[i];
        }
        const _Float16* wb = Wswz + (((size_t)kc * 64 + jt) * 32 + lane) * 16;
        acc0 = wmma_f16(a, *(const v16h*)(wb + 0 * 8192), acc0);  // gate i
        acc1 = wmma_f16(a, *(const v16h*)(wb + 1 * 8192), acc1);  // gate f
        acc2 = wmma_f16(a, *(const v16h*)(wb + 2 * 8192), acc2);  // gate g
        acc3 = wmma_f16(a, *(const v16h*)(wb + 3 * 8192), acc3);  // gate o
    }

    // ---- h contribution: K chunks 2..9 (K = 64..319), h already f16
#pragma unroll
    for (int kc = 2; kc < 10; ++kc) {
        const int Kh = (kc - 2) * 32 + kb;
        const _Float16* hp = h_in + (size_t)row * H_ + Kh;
        v8h hA = *(const v8h*)hp;
        v8h hB = *(const v8h*)(hp + 16);
        v16h a;
#pragma unroll
        for (int i = 0; i < 8; ++i) {
            a[i]     = hA[i];
            a[8 + i] = hB[i];
        }
        const _Float16* wb = Wswz + (((size_t)kc * 64 + jt) * 32 + lane) * 16;
        acc0 = wmma_f16(a, *(const v16h*)(wb + 0 * 8192), acc0);
        acc1 = wmma_f16(a, *(const v16h*)(wb + 1 * 8192), acc1);
        acc2 = wmma_f16(a, *(const v16h*)(wb + 2 * 8192), acc2);
        acc3 = wmma_f16(a, *(const v16h*)(wb + 3 * 8192), acc3);
    }

    // ---- fused cell update. D layout: lane -> col j0+(lane&15);
    //      vgpr r -> row m0 + r + (lane>=16 ? 8 : 0)
    const int j  = j0 + (lane & 15);
    const float bi = bsum[j];
    const float bf = bsum[H_ + j];
    const float bg = bsum[2 * H_ + j];
    const float bo = bsum[3 * H_ + j];
    const int mhi = (lane >> 4) << 3;
#pragma unroll
    for (int r = 0; r < 8; ++r) {
        const int m = m0 + mhi + r;
        const size_t off = (size_t)m * H_ + j;
        float gi = acc0[r] + bi;
        float gf = acc1[r] + bf;
        float gg = acc2[r] + bg;
        float go = acc3[r] + bo;
        float cn = sigf(gf) * c[off] + sigf(gi) * tanhfast(gg);
        c[off] = cn;
        h_out[off] = (_Float16)(sigf(go) * tanhfast(cn));
    }
}

// ---------------------------------------------------------------------------
// Final projection: out[512,128] = h_fin @ w_out^T + b_out, f16 WMMA.
// Grid: 64 blocks x 128 threads = 256 waves = 32 m-tiles x 8 n-tiles.
// ---------------------------------------------------------------------------
__global__ __launch_bounds__(128) void out_proj(const _Float16* __restrict__ h,
                                                const float* __restrict__ w_out,
                                                const float* __restrict__ b_out,
                                                float* __restrict__ out) {
    const int lane = threadIdx.x & 31;
    const int wid  = (blockIdx.x * blockDim.x + threadIdx.x) >> 5;  // 0..255
    const int mt   = wid & 31;
    const int nt   = wid >> 5;          // 0..7
    const int m0   = mt * 16;
    const int rowA = m0 + (lane & 15);
    const int nB   = nt * 16 + (lane & 15);
    const int kb   = (lane < 16) ? 0 : 8;

    v8f acc = {};
#pragma unroll
    for (int kc = 0; kc < 8; ++kc) {
        const int K0 = kc * 32 + kb;
        const _Float16* hp = h + (size_t)rowA * H_ + K0;
        v8h hA = *(const v8h*)hp;
        v8h hB = *(const v8h*)(hp + 16);
        v16h a;
#pragma unroll
        for (int i = 0; i < 8; ++i) {
            a[i]     = hA[i];
            a[8 + i] = hB[i];
        }
        // B(K,N) = w_out[N*H + K]: per-lane contiguous in K -> convert f32->f16
        const float* wp = w_out + (size_t)nB * H_ + K0;
        v4f w0 = *(const v4f*)wp;
        v4f w1 = *(const v4f*)(wp + 4);
        v4f w2 = *(const v4f*)(wp + 16);
        v4f w3 = *(const v4f*)(wp + 20);
        v16h b;
#pragma unroll
        for (int i = 0; i < 4; ++i) {
            b[i]      = (_Float16)w0[i];
            b[4 + i]  = (_Float16)w1[i];
            b[8 + i]  = (_Float16)w2[i];
            b[12 + i] = (_Float16)w3[i];
        }
        acc = wmma_f16(a, b, acc);
    }

    const float bb = b_out[nB];
    const int mhi = (lane >> 4) << 3;
#pragma unroll
    for (int r = 0; r < 8; ++r) {
        out[(size_t)(m0 + mhi + r) * OUT_ + nB] = acc[r] + bb;
    }
}

// ---------------------------------------------------------------------------
extern "C" void kernel_launch(void* const* d_in, const int* in_sizes, int n_in,
                              void* d_out, int out_size, void* d_ws, size_t ws_size,
                              hipStream_t stream) {
    (void)in_sizes; (void)n_in; (void)out_size; (void)ws_size;
    const float* x     = (const float*)d_in[0];
    const float* w_ih  = (const float*)d_in[1];
    const float* w_hh  = (const float*)d_in[2];
    const float* b_ih  = (const float*)d_in[3];
    const float* b_hh  = (const float*)d_in[4];
    const float* w_out = (const float*)d_in[5];
    const float* b_out = (const float*)d_in[6];
    float* out = (float*)d_out;

    // Workspace layout (~1.7 MB total)
    float*    c    = (float*)d_ws;                     // 512*256 f32
    _Float16* hbuf = (_Float16*)(c + B_ * H_);         // 2 x 512*256 f16 (ping-pong)
    _Float16* Wswz = hbuf + 2 * B_ * H_;               // 320*1024 f16, fragment order
    float*    bsum = (float*)(Wswz + 10 * 64 * 32 * 16);  // 1024 f32

    init_state<<<(B_ * H_ + 255) / 256, 256, 0, stream>>>(c, hbuf, b_ih, b_hh, bsum);
    swizzle_w<<<(10 * 64 * 32 * 16) / 256, 256, 0, stream>>>(w_ih, w_hh, Wswz);

    // Reversed-time scan: step s uses original timestep t = T-1-s.
    for (int s = 0; s < T_; ++s) {
        const int t = T_ - 1 - s;
        const _Float16* hin = hbuf + (size_t)(s & 1) * B_ * H_;
        _Float16* hout      = hbuf + (size_t)((s + 1) & 1) * B_ * H_;
        lstm_step<<<128, 128, 0, stream>>>(x, Wswz, bsum, hin, hout, c, t);
    }

    // T_ is even -> final h lives in buffer (T_ & 1) == 0.
    out_proj<<<64, 128, 0, stream>>>(hbuf + (size_t)(T_ & 1) * B_ * H_,
                                     w_out, b_out, out);
}
